// BlockLayer_64063732187161
// MI455X (gfx1250) — matrix operand order repaired and checked by reading, not compiled, not run on hardware
//
#include <hip/hip_runtime.h>
#include <hip/hip_bf16.h>
#include <stdint.h>

#define Bsz  8
#define Tn   1024
#define Dm   1024
#define Hn   16
#define En   8
#define HSz  64
#define Ntok (Bsz * Tn)   /* 8192 */
#define D4   (4 * Dm)     /* 4096 */

typedef __attribute__((ext_vector_type(16))) _Float16 v16h;
typedef __attribute__((ext_vector_type(8)))  _Float16 v8h;
typedef __attribute__((ext_vector_type(8)))  float    v8f;
typedef __attribute__((ext_vector_type(4)))  unsigned int v4u;
typedef __attribute__((ext_vector_type(8)))  int      v8i;
typedef __attribute__((ext_vector_type(4)))  int      v4i;

#if __has_builtin(__builtin_amdgcn_tensor_load_to_lds)
#define HAVE_TDM 1
#else
#define HAVE_TDM 0
#endif

static __device__ __forceinline__ v16h cat8(v8h lo, v8h hi) {
  return __builtin_shufflevector(lo, hi, 0,1,2,3,4,5,6,7,8,9,10,11,12,13,14,15);
}

static __device__ __forceinline__ v8f wmma_f16(v16h a, v16h b, v8f c) {
  return __builtin_amdgcn_wmma_f32_16x16x32_f16(false, a, false, b, (short)0, c, false, false);
}

// One 16x16 fp32 tile streamed straight from global (used for QKV / attention).
static __device__ __forceinline__ v8f wmma_tile(const _Float16* __restrict__ A, int lda,
                                                const _Float16* __restrict__ BT, int ldb,
                                                int Kd, int lane, v8f acc) {
  const int mr = lane & 15;
  const int hs = lane >> 4;
  const _Float16* arow = A  + (size_t)mr * lda + (hs << 3);
  const _Float16* brow = BT + (size_t)mr * ldb + (hs << 3);
  for (int k0 = 0; k0 < Kd; k0 += 32) {
    v8h  alo = *(const v8h*)(arow + k0);
    v8h  ahi = *(const v8h*)(arow + k0 + 16);
    v16h a   = cat8(alo, ahi);
    v16h b   = cat8(*(const v8h*)(brow + k0), *(const v8h*)(brow + k0 + 16));
    __builtin_prefetch(brow + k0 + 512, 0, 1);
    acc = wmma_f16(a, b, acc);
  }
  return acc;
}

// ---------------- TDM: 2D tile (128 rows x 32 halves) global -> LDS ----------------
// D# per CDNA5 ISA ch.8. pad_enable inserts 4 DWORDs after every 16 DWORDs, i.e. LDS
// row stride = 32 + 8 halves = LDST, matching the compute-side layout.
#define BM   128
#define BN   128
#define BKc  32
#define LDST 40

static __device__ __forceinline__ void tdm_load_tile(uint32_t lds_byte_addr,
                                                     const _Float16* gsrc,
                                                     uint32_t stride_elems) {
#if HAVE_TDM
  const uint64_t ga = (uint64_t)(uintptr_t)gsrc;
  v4u g0;
  g0[0] = 1u;                                        // count = 1 valid descriptor
  g0[1] = lds_byte_addr;                             // lds_addr [63:32]
  g0[2] = (unsigned)(ga & 0xFFFFFFFFu);              // global_addr [95:64]
  g0[3] = (unsigned)((ga >> 32) & 0x01FFFFFFu) | 0x80000000u;  // addr[56:32] | type=2
  v8i g1;
  g1[0] = (int)((1u << 16) | (1u << 20) | (3u << 22) | (3u << 25)); // 2B elems, pad 4DW/16DW
  g1[1] = (int)((stride_elems & 0xFFFFu) << 16);                    // tensor_dim0 lo16
  g1[2] = (int)(((stride_elems >> 16) & 0xFFFFu) | (0x4000u << 16)); // dim0 hi16 | dim1=16384 lo16
  g1[3] = (int)(32u << 16);                                         // dim1 hi16=0 | tile_dim0=32
  g1[4] = (int)128u;                                                // tile_dim1=128, tile_dim2=0
  g1[5] = (int)stride_elems;                                        // dim0_stride lo32
  g1[6] = 0;
  g1[7] = 0;
  v4i z4 = {0, 0, 0, 0};
#if defined(__clang_major__) && __clang_major__ >= 23
  v8i z8 = {0, 0, 0, 0, 0, 0, 0, 0};
  __builtin_amdgcn_tensor_load_to_lds(g0, g1, z4, z4, z8, 0);
#else
  __builtin_amdgcn_tensor_load_to_lds(g0, g1, z4, z4, 0);
#endif
#else
  (void)lds_byte_addr; (void)gsrc; (void)stride_elems;
#endif
}

// ---------------- conversion / transpose ----------------

__global__ void cvt_f16_kernel(const float* __restrict__ in, _Float16* __restrict__ out, int n) {
  const int i = blockIdx.x * blockDim.x + threadIdx.x;
  if (i < n) out[i] = (_Float16)in[i];
}

// in: [mats][K][Nn] fp32 -> out: [mats][Nn][K] f16 (BT layout)
__global__ void transpose_cvt_kernel(const float* __restrict__ in, _Float16* __restrict__ out,
                                     int K, int Nn) {
  const int mat = blockIdx.z;
  const int idx = blockIdx.x * blockDim.x + threadIdx.x;
  if (idx >= K * Nn) return;
  const int k = idx / Nn, n = idx % Nn;
  const size_t base = (size_t)mat * K * Nn;
  out[base + (size_t)n * K + k] = (_Float16)in[base + (size_t)k * Nn + n];
}

__global__ void zero_kernel(float* __restrict__ p, int n) {
  const int i = blockIdx.x * blockDim.x + threadIdx.x;
  if (i < n) p[i] = 0.f;
}

// ---------------- QKV projection (WMMA GEMM) ----------------
__global__ void qkv_kernel(const _Float16* __restrict__ xh, const _Float16* __restrict__ wT,
                           _Float16* __restrict__ q, _Float16* __restrict__ kk,
                           _Float16* __restrict__ vT) {
  const int lane  = threadIdx.x;
  const int mt    = blockIdx.x * blockDim.y + threadIdx.y;
  const int nt    = blockIdx.y;
  const int which = blockIdx.z;
  const _Float16* A  = xh + (size_t)mt * 16 * Dm;
  const _Float16* BT = wT + (size_t)which * Dm * Dm + (size_t)nt * 16 * Dm;
  v8f acc = {};
  acc = wmma_tile(A, Dm, BT, Dm, Dm, lane, acc);
  const int cn = lane & 15;
  const int rb = (lane >> 4) * 8;
  const int nn = nt * 16 + cn;
  const int h = nn >> 6, ee = nn & 63;
#pragma unroll
  for (int r = 0; r < 8; ++r) {
    const int m = mt * 16 + rb + r;
    const int b = m >> 10, t = m & (Tn - 1);
    const int bh = b * Hn + h;
    const _Float16 val = (_Float16)acc[r];
    if (which == 0)      q [((size_t)bh * Tn + t) * HSz + ee] = val;
    else if (which == 1) kk[((size_t)bh * Tn + t) * HSz + ee] = val;
    else                 vT[((size_t)bh * HSz + ee) * Tn + t] = val;
  }
}

// ---------------- flash attention (one wave per 16-query tile) ----------------
__global__ void attn_kernel(const _Float16* __restrict__ q, const _Float16* __restrict__ k,
                            const _Float16* __restrict__ vT, float* __restrict__ attnO) {
  __shared__ __align__(32) _Float16 p_lds[16 * 32];
  const int lane = threadIdx.x;
  const int qt = blockIdx.x;
  const int bh = blockIdx.y;
  const int t0 = qt * 16;
  const int mr = lane & 15;
  const int hs = lane >> 4;
  const int cn = mr;
  const int rows_lo = hs * 8;
  const float SCALE = 0.03125f;     // D^-0.5
  const float NEGB  = -1.0e30f;

  const _Float16* qb = q  + (size_t)bh * Tn * HSz;
  const _Float16* kb = k  + (size_t)bh * Tn * HSz;
  const _Float16* vb = vT + (size_t)bh * HSz * Tn;

  v16h qa[2];
#pragma unroll
  for (int c = 0; c < 2; ++c) {
    const _Float16* qrow = qb + (size_t)(t0 + mr) * HSz + c * 32 + (hs << 3);
    qa[c] = cat8(*(const v8h*)qrow, *(const v8h*)(qrow + 16));
  }

  v8f o[4] = {v8f{}, v8f{}, v8f{}, v8f{}};
  float rmax[8], rsum[8];
#pragma unroll
  for (int r = 0; r < 8; ++r) { rmax[r] = NEGB; rsum[r] = 0.f; }

  for (int s0 = 0; s0 <= t0 + 15; s0 += 32) {
    v8f sc0 = {}, sc1 = {};
    const _Float16* k0p = kb + (size_t)(s0 + mr) * HSz + (hs << 3);
    const _Float16* k1p = kb + (size_t)(s0 + 16 + mr) * HSz + (hs << 3);
#pragma unroll
    for (int c = 0; c < 2; ++c) {
      sc0 = wmma_f16(qa[c], cat8(*(const v8h*)(k0p + c * 32), *(const v8h*)(k0p + c * 32 + 16)), sc0);
      sc1 = wmma_f16(qa[c], cat8(*(const v8h*)(k1p + c * 32), *(const v8h*)(k1p + c * 32 + 16)), sc1);
    }
#pragma unroll
    for (int r = 0; r < 8; ++r) {
      const int trow = t0 + rows_lo + r;
      float v0 = sc0[r] * SCALE;
      float v1 = sc1[r] * SCALE;
      if (s0 + cn > trow)      v0 = NEGB;
      if (s0 + 16 + cn > trow) v1 = NEGB;
      float mx = fmaxf(v0, v1);
#pragma unroll
      for (int off = 8; off >= 1; off >>= 1) mx = fmaxf(mx, __shfl_xor(mx, off, 32));
      const float nm   = fmaxf(rmax[r], mx);
      const float corr = __expf(rmax[r] - nm);
      const float p0   = __expf(v0 - nm);
      const float p1   = __expf(v1 - nm);
      float rs = p0 + p1;
#pragma unroll
      for (int off = 8; off >= 1; off >>= 1) rs += __shfl_xor(rs, off, 32);
      rsum[r] = rsum[r] * corr + rs;
      rmax[r] = nm;
#pragma unroll
      for (int nt = 0; nt < 4; ++nt) o[nt][r] *= corr;
      p_lds[(rows_lo + r) * 32 + cn]      = (_Float16)p0;
      p_lds[(rows_lo + r) * 32 + 16 + cn] = (_Float16)p1;
    }
    asm volatile("s_wait_dscnt 0" ::: "memory");
    v8h  plo = *(const v8h*)(p_lds + mr * 32 + (hs << 3));
    v8h  phi = *(const v8h*)(p_lds + mr * 32 + 16 + (hs << 3));
    v16h pa  = cat8(plo, phi);
#pragma unroll
    for (int nt = 0; nt < 4; ++nt) {
      const _Float16* vrow = vb + (size_t)(nt * 16 + mr) * Tn + s0 + (hs << 3);
      o[nt] = wmma_f16(pa, cat8(*(const v8h*)vrow, *(const v8h*)(vrow + 16)), o[nt]);
    }
  }

  const int b = bh >> 4, h = bh & 15;
#pragma unroll
  for (int nt = 0; nt < 4; ++nt) {
    const int col = h * HSz + nt * 16 + cn;
#pragma unroll
    for (int r = 0; r < 8; ++r) {
      const int t = t0 + rows_lo + r;
      attnO[((size_t)b * Tn + t) * Dm + col] = o[nt][r] / rsum[r];
    }
  }
}

// ---------------- LayerNorm + residual ----------------
__global__ void ln_residual_kernel(const float* __restrict__ resid, const float* __restrict__ hin,
                                   const float* __restrict__ g, const float* __restrict__ bb,
                                   float* __restrict__ out) {
  __shared__ float red[256];
  const int row = blockIdx.x;
  const int tid = threadIdx.x;
  const float* hr = hin + (size_t)row * Dm;
  float v[4];
  float s = 0.f;
#pragma unroll
  for (int i = 0; i < 4; ++i) { v[i] = hr[tid + i * 256]; s += v[i]; }
  red[tid] = s; __syncthreads();
  for (int o = 128; o > 0; o >>= 1) { if (tid < o) red[tid] += red[tid + o]; __syncthreads(); }
  const float mu = red[0] * (1.f / Dm);
  __syncthreads();
  float s2 = 0.f;
#pragma unroll
  for (int i = 0; i < 4; ++i) { const float d = v[i] - mu; s2 += d * d; }
  red[tid] = s2; __syncthreads();
  for (int o = 128; o > 0; o >>= 1) { if (tid < o) red[tid] += red[tid + o]; __syncthreads(); }
  const float rstd = rsqrtf(red[0] * (1.f / Dm) + 1e-5f);
#pragma unroll
  for (int i = 0; i < 4; ++i) {
    const int c = tid + i * 256;
    out[(size_t)row * Dm + c] = resid[(size_t)row * Dm + c] + (v[i] - mu) * rstd * g[c] + bb[c];
  }
}

// ---------------- MoE gating ----------------
__global__ void gate_logits_kernel(const float* __restrict__ x, const float* __restrict__ Wg,
                                   float* __restrict__ logits) {
  const int tid = blockIdx.x * blockDim.x + threadIdx.x;
  if (tid >= Ntok * En) return;
  const int t = tid >> 3, e = tid & 7;
  const float* xr = x + (size_t)t * Dm;
  float s = 0.f;
  for (int d = 0; d < Dm; ++d) s += xr[d] * Wg[d * En + e];
  logits[tid] = s;
}

__global__ void gate_topk_kernel(const float* __restrict__ logits, float* __restrict__ we) {
  const int t = blockIdx.x * blockDim.x + threadIdx.x;
  if (t >= Ntok) return;
  float l[En];
#pragma unroll
  for (int j = 0; j < En; ++j) l[j] = logits[t * En + j];
  int i0 = 0;
#pragma unroll
  for (int j = 1; j < En; ++j) if (l[j] > l[i0]) i0 = j;
  int i1 = (i0 == 0) ? 1 : 0;
#pragma unroll
  for (int j = 0; j < En; ++j) if (j != i0 && l[j] > l[i1]) i1 = j;
  const float m  = fmaxf(l[i0], l[i1]);
  const float e0 = __expf(l[i0] - m), e1 = __expf(l[i1] - m);
  const float inv = 1.f / (e0 + e1);
#pragma unroll
  for (int j = 0; j < En; ++j) {
    float w = 0.f;
    if (j == i0) w = e0 * inv;
    if (j == i1) w = e1 * inv;
    we[t * En + j] = w;
  }
}

// ---------------- MoE expert GEMMs: 128x128 block tile, TDM-fed, double-buffered ----
// 8 waves, each owning a 32x64 sub-tile (2x4 grid of 16x16 WMMA tiles).
// MODE 0: outH = relu(A@B + bias) as f16.  MODE 1: outF += we[:,e]*(A@B + bias).
template <int MODE>
__global__ void moe_gemm_tiled(const _Float16* __restrict__ A, const _Float16* __restrict__ BT,
                               const float* __restrict__ bias, const float* __restrict__ wegate,
                               int e, int Kd,
                               _Float16* __restrict__ outH, float* __restrict__ outF) {
  __shared__ __align__(16) _Float16 lds_a[2][BM * LDST];
  __shared__ __align__(16) _Float16 lds_b[2][BN * LDST];
  const int lane = threadIdx.x;
  const int wave = threadIdx.y;       // 0..7
  const int wm = wave >> 1;           // 0..3 -> m offset wm*32
  const int wn = wave & 1;            // 0..1 -> n offset wn*64
  const int mr = lane & 15, hs = lane >> 4;
  const int m0 = blockIdx.x * BM;
  const int n0 = blockIdx.y * BN;
  const _Float16* Ag = A  + (size_t)m0 * Kd;
  const _Float16* Bg = BT + (size_t)n0 * Kd;

  v8f acc[2][4];
#pragma unroll
  for (int i = 0; i < 2; ++i)
#pragma unroll
    for (int j = 0; j < 4; ++j) acc[i][j] = v8f{};

  const int nk = Kd / BKc;
#if HAVE_TDM
  if (wave == 0) {
    tdm_load_tile((uint32_t)(uintptr_t)&lds_a[0][0], Ag, (uint32_t)Kd);
    tdm_load_tile((uint32_t)(uintptr_t)&lds_b[0][0], Bg, (uint32_t)Kd);
  }
#endif
  for (int i = 0; i < nk; ++i) {
    const int cur = i & 1;
#if HAVE_TDM
    if (wave == 0) {
      if (i + 1 < nk) {
        tdm_load_tile((uint32_t)(uintptr_t)&lds_a[cur ^ 1][0], Ag + (size_t)(i + 1) * BKc, (uint32_t)Kd);
        tdm_load_tile((uint32_t)(uintptr_t)&lds_b[cur ^ 1][0], Bg + (size_t)(i + 1) * BKc, (uint32_t)Kd);
        __builtin_amdgcn_s_wait_tensorcnt(2);   // in-order TDM: stage i pair complete
      } else {
        __builtin_amdgcn_s_wait_tensorcnt(0);
      }
    }
    __syncthreads();
#else
    {
      const int tid = wave * 32 + lane;
      const _Float16* ga = Ag + (size_t)i * BKc;
      const _Float16* gb = Bg + (size_t)i * BKc;
      for (int idx = tid; idx < BM * 4; idx += 256) {
        const int row = idx >> 2, c = (idx & 3) * 8;
        *(v8h*)&lds_a[cur][row * LDST + c] = *(const v8h*)(ga + (size_t)row * Kd + c);
        *(v8h*)&lds_b[cur][row * LDST + c] = *(const v8h*)(gb + (size_t)row * Kd + c);
      }
      __syncthreads();
    }
#endif
    v16h af[2], bf[4];
#pragma unroll
    for (int sm = 0; sm < 2; ++sm) {
      const _Float16* ap = &lds_a[cur][(wm * 32 + sm * 16 + mr) * LDST + hs * 8];
      af[sm] = cat8(*(const v8h*)ap, *(const v8h*)(ap + 16));
    }
#pragma unroll
    for (int sn = 0; sn < 4; ++sn) {
      const _Float16* bp = &lds_b[cur][(wn * 64 + sn * 16 + mr) * LDST + hs * 8];
      bf[sn] = cat8(*(const v8h*)bp, *(const v8h*)(bp + 16));
    }
#pragma unroll
    for (int sm = 0; sm < 2; ++sm)
#pragma unroll
      for (int sn = 0; sn < 4; ++sn)
        acc[sm][sn] = wmma_f16(af[sm], bf[sn], acc[sm][sn]);
    __syncthreads();
  }

  const int cn = lane & 15, rb = hs * 8;
#pragma unroll
  for (int sm = 0; sm < 2; ++sm) {
#pragma unroll
    for (int sn = 0; sn < 4; ++sn) {
      const int n = n0 + wn * 64 + sn * 16 + cn;
      const float bv = bias[n];
#pragma unroll
      for (int r = 0; r < 8; ++r) {
        const int m = m0 + wm * 32 + sm * 16 + rb + r;
        const float v = acc[sm][sn][r] + bv;
        if constexpr (MODE == 0) {
          outH[(size_t)m * D4 + n] = (_Float16)fmaxf(v, 0.f);
        } else {
          outF[(size_t)m * Dm + n] += wegate[m * En + e] * v;
        }
      }
    }
  }
}

// ---------------- host ----------------
extern "C" void kernel_launch(void* const* d_in, const int* in_sizes, int n_in,
                              void* d_out, int out_size, void* d_ws, size_t ws_size,
                              hipStream_t stream) {
  (void)in_sizes; (void)n_in; (void)out_size; (void)ws_size;
  const float* x   = (const float*)d_in[0];
  const float* Wq  = (const float*)d_in[1];
  const float* Wk  = (const float*)d_in[2];
  const float* Wv  = (const float*)d_in[3];
  const float* Wg  = (const float*)d_in[4];
  const float* W1  = (const float*)d_in[5];
  const float* b1  = (const float*)d_in[6];
  const float* W2  = (const float*)d_in[7];
  const float* b2  = (const float*)d_in[8];
  const float* g1  = (const float*)d_in[9];
  const float* be1 = (const float*)d_in[10];
  const float* g2  = (const float*)d_in[11];
  const float* be2 = (const float*)d_in[12];
  float* out = (float*)d_out;

  char* p = (char*)d_ws;
  auto take = [&](size_t bytes) -> char* {
    char* r = p; p += (bytes + 255) & ~(size_t)255; return r;
  };
  _Float16* xh     = (_Float16*)take((size_t)Ntok * Dm * 2);
  _Float16* wqkvT  = (_Float16*)take((size_t)3 * Dm * Dm * 2);
  _Float16* qbuf   = (_Float16*)take((size_t)Ntok * Dm * 2);
  _Float16* kbuf   = (_Float16*)take((size_t)Ntok * Dm * 2);
  _Float16* vTbuf  = (_Float16*)take((size_t)Ntok * Dm * 2);
  float*    attnO  = (float*)take((size_t)Ntok * Dm * 4);
  _Float16* w1T    = (_Float16*)take((size_t)En * Dm * D4 * 2);
  _Float16* w2T    = (_Float16*)take((size_t)En * Dm * D4 * 2);
  _Float16* hbuf   = (_Float16*)take((size_t)Ntok * D4 * 2);
  float*    moeO   = (float*)take((size_t)Ntok * Dm * 4);
  float*    logits = (float*)take((size_t)Ntok * En * 4);
  float*    wegate = (float*)take((size_t)Ntok * En * 4);

  cvt_f16_kernel<<<(Ntok * Dm) / 256, 256, 0, stream>>>(x, xh, Ntok * Dm);
  transpose_cvt_kernel<<<dim3((Dm * HSz) / 256, 1, Hn), 256, 0, stream>>>(Wq, wqkvT,               Dm, HSz);
  transpose_cvt_kernel<<<dim3((Dm * HSz) / 256, 1, Hn), 256, 0, stream>>>(Wk, wqkvT + Dm * Dm,     Dm, HSz);
  transpose_cvt_kernel<<<dim3((Dm * HSz) / 256, 1, Hn), 256, 0, stream>>>(Wv, wqkvT + 2 * Dm * Dm, Dm, HSz);
  transpose_cvt_kernel<<<dim3((Dm * D4) / 256, 1, En), 256, 0, stream>>>(W1, w1T, Dm, D4);
  transpose_cvt_kernel<<<dim3((D4 * Dm) / 256, 1, En), 256, 0, stream>>>(W2, w2T, D4, Dm);

  // attention branch
  qkv_kernel<<<dim3(Ntok / 64, Dm / 16, 3), dim3(32, 4), 0, stream>>>(xh, wqkvT, qbuf, kbuf, vTbuf);
  attn_kernel<<<dim3(Tn / 16, Bsz * Hn), 32, 0, stream>>>(qbuf, kbuf, vTbuf, attnO);
  ln_residual_kernel<<<Ntok, 256, 0, stream>>>(x, attnO, g1, be1, out);

  // MoE branch (from original x)
  gate_logits_kernel<<<(Ntok * En) / 256, 256, 0, stream>>>(x, Wg, logits);
  gate_topk_kernel<<<Ntok / 256, 256, 0, stream>>>(logits, wegate);
  zero_kernel<<<(Ntok * Dm) / 256, 256, 0, stream>>>(moeO, Ntok * Dm);
  for (int e = 0; e < En; ++e) {
    moe_gemm_tiled<0><<<dim3(Ntok / BM, D4 / BN), dim3(32, 8), 0, stream>>>(
        xh, w1T + (size_t)e * Dm * D4, b1 + (size_t)e * D4, nullptr, e, Dm, hbuf, nullptr);
    moe_gemm_tiled<1><<<dim3(Ntok / BM, Dm / BN), dim3(32, 8), 0, stream>>>(
        hbuf, w2T + (size_t)e * Dm * D4, b2 + (size_t)e * Dm, wegate, e, D4, nullptr, moeO);
  }
  ln_residual_kernel<<<Ntok, 256, 0, stream>>>(out, moeO, g2, be2, out);
}
